// ConditionalAttentionConv2D_20641612824790
// MI455X (gfx1250) — matrix operand order repaired and checked by reading, not compiled, not run on hardware
//
#include <hip/hip_runtime.h>
#include <hip/hip_bf16.h>

typedef __bf16 bf16_t;
typedef __attribute__((ext_vector_type(16))) __bf16 v16bf;
typedef __attribute__((ext_vector_type(8)))  __bf16 v8bf;
typedef __attribute__((ext_vector_type(4)))  __bf16 v4bf;
typedef __attribute__((ext_vector_type(8)))  float  v8f;

#define B_   8
#define H_   48
#define W_   192
#define C_   128
#define DK_  16
#define PH_  24
#define PW_  96
#define NQ_  (H_*W_)    // 9216
#define NK_  (PH_*PW_)  // 2304
#define QTILES (NQ_/16) // 576
#define KB32  (NK_/32)  // 72

__device__ __forceinline__ v8f wmma_bf16(v16bf a, v16bf b, v8f c) {
  return __builtin_amdgcn_wmma_f32_16x16x32_bf16(false, a, false, b, (short)0, c, false, false);
}

// ---- A-matrix tile from fp32 pixel row (16x32 bf16, K = channels block) ----
__device__ __forceinline__ void load_a_row(const float* xrow, int hi, v16bf A[4]) {
#pragma unroll
  for (int kb = 0; kb < 4; kb++) {
    int cb = kb*32 + (hi ? 8 : 0);
    float4 f0 = *(const float4*)(xrow + cb);
    float4 f1 = *(const float4*)(xrow + cb + 4);
    float4 f2 = *(const float4*)(xrow + cb + 16);
    float4 f3 = *(const float4*)(xrow + cb + 20);
    v16bf a;
    a[0]=(bf16_t)f0.x;  a[1]=(bf16_t)f0.y;  a[2]=(bf16_t)f0.z;  a[3]=(bf16_t)f0.w;
    a[4]=(bf16_t)f1.x;  a[5]=(bf16_t)f1.y;  a[6]=(bf16_t)f1.z;  a[7]=(bf16_t)f1.w;
    a[8]=(bf16_t)f2.x;  a[9]=(bf16_t)f2.y;  a[10]=(bf16_t)f2.z; a[11]=(bf16_t)f2.w;
    a[12]=(bf16_t)f3.x; a[13]=(bf16_t)f3.y; a[14]=(bf16_t)f3.z; a[15]=(bf16_t)f3.w;
    A[kb] = a;
  }
}

// ---- B-matrix tile from transposed bf16 weights (32x16, lane = out-column) ----
__device__ __forceinline__ void load_b_wt(const bf16_t* wrow, int hi, v16bf Bw[4]) {
#pragma unroll
  for (int kb = 0; kb < 4; kb++) {
    int cb = kb*32 + (hi ? 8 : 0);
    v8bf lo  = *(const v8bf*)(wrow + cb);
    v8bf hi8 = *(const v8bf*)(wrow + cb + 16);
    v16bf bb;
#pragma unroll
    for (int i = 0; i < 8; i++) { bb[i] = lo[i]; bb[8+i] = hi8[i]; }
    Bw[kb] = bb;
  }
}

// ================= kernel 0: weight transpose + bf16 convert =================
__global__ void prep_weights_k(const float* __restrict__ Wq, const float* __restrict__ Wk,
                               const float* __restrict__ Wv,
                               bf16_t* __restrict__ WqT, bf16_t* __restrict__ WkT,
                               bf16_t* __restrict__ WvT) {
  int i = blockIdx.x * 256 + threadIdx.x;
  if (i < C_*DK_) {
    int c = i / DK_, d = i % DK_;
    WqT[d*C_ + c] = (bf16_t)Wq[i];
    WkT[d*C_ + c] = (bf16_t)Wk[i];
  }
  if (i < C_*C_) {
    int c = i / C_, cp = i % C_;
    WvT[cp*C_ + c] = (bf16_t)Wv[i];
  }
}

// ====== kernel 1: projections + bias + 2x2 maxpool (WMMA per row-pair) =======
// One wave handles pixel rows (2*h2, 2*h2+1), cols w0*16..+15.
__global__ __launch_bounds__(32)
void stageA_k(const float* __restrict__ query,
              const float* __restrict__ bq, const float* __restrict__ bk,
              const float* __restrict__ bv,
              const bf16_t* __restrict__ WqT, const bf16_t* __restrict__ WkT,
              const bf16_t* __restrict__ WvT,
              bf16_t* __restrict__ qb, bf16_t* __restrict__ kp, bf16_t* __restrict__ vt) {
  int gid = blockIdx.x;
  int b   = gid / (PH_ * 12);
  int rem = gid % (PH_ * 12);
  int h2  = rem / 12, w0 = rem % 12;
  int lane = threadIdx.x;
  int lp = lane & 15;
  int hi = (lane >= 16) ? 1 : 0;

  int h0 = 2*h2, h1 = h0 + 1;
  const float* x0 = query + ((size_t)b*NQ_ + (size_t)h0*W_ + w0*16 + lp) * C_;
  const float* x1 = query + ((size_t)b*NQ_ + (size_t)h1*W_ + w0*16 + lp) * C_;
  v16bf A0[4], A1[4];
  load_a_row(x0, hi, A0);
  load_a_row(x1, hi, A1);

  int kbase = h2*PW_ + w0*8;  // first pooled key this wave produces

  // ---- V projection (8 chunks of 16 out-channels), pooled, stored as V^T ----
#pragma unroll 1
  for (int j = 0; j < 8; j++) {
    v16bf Bw[4];
    load_b_wt(WvT + (size_t)(j*16 + lp)*C_, hi, Bw);
    v8f a0 = {}, a1 = {};
#pragma unroll
    for (int kb = 0; kb < 4; kb++) { a0 = wmma_bf16(A0[kb], Bw[kb], a0); a1 = wmma_bf16(A1[kb], Bw[kb], a1); }
    float bias = bv[j*16 + lp];
    v4bf pk;
#pragma unroll
    for (int p = 0; p < 4; p++) {
      float m = fmaxf(fmaxf(a0[2*p], a0[2*p+1]), fmaxf(a1[2*p], a1[2*p+1]));
      pk[p] = (bf16_t)(m + bias);
    }
    *(v4bf*)(vt + ((size_t)b*C_ + j*16 + lp)*NK_ + kbase + hi*4) = pk;
  }

  // ---- K projection (DK=16), pooled, row-major [key][d] ----
  {
    v16bf Bw[4];
    load_b_wt(WkT + (size_t)lp*C_, hi, Bw);
    v8f a0 = {}, a1 = {};
#pragma unroll
    for (int kb = 0; kb < 4; kb++) { a0 = wmma_bf16(A0[kb], Bw[kb], a0); a1 = wmma_bf16(A1[kb], Bw[kb], a1); }
    float bias = bk[lp];
#pragma unroll
    for (int p = 0; p < 4; p++) {
      float m = fmaxf(fmaxf(a0[2*p], a0[2*p+1]), fmaxf(a1[2*p], a1[2*p+1]));
      kp[((size_t)b*NK_ + kbase + hi*4 + p)*DK_ + lp] = (bf16_t)(m + bias);
    }
  }

  // ---- Q projection (no pooling), row-major [pix][d] ----
  {
    v16bf Bw[4];
    load_b_wt(WqT + (size_t)lp*C_, hi, Bw);
    v8f a0 = {}, a1 = {};
#pragma unroll
    for (int kb = 0; kb < 4; kb++) { a0 = wmma_bf16(A0[kb], Bw[kb], a0); a1 = wmma_bf16(A1[kb], Bw[kb], a1); }
    float bias = bq[lp];
    size_t p0 = (size_t)b*NQ_ + (size_t)h0*W_ + w0*16 + hi*8;
    size_t p1 = (size_t)b*NQ_ + (size_t)h1*W_ + w0*16 + hi*8;
#pragma unroll
    for (int r = 0; r < 8; r++) {
      qb[(p0 + r)*DK_ + lp] = (bf16_t)(a0[r] + bias);
      qb[(p1 + r)*DK_ + lp] = (bf16_t)(a1[r] + bias);
    }
  }
}

// ========== kernel 2: fused attention, one wave per 16-query tile ===========
__global__ __launch_bounds__(32)
void stageB_k(const float* __restrict__ query, const bf16_t* __restrict__ qb,
              const bf16_t* __restrict__ kp, const bf16_t* __restrict__ vt,
              const float* __restrict__ beta, float* __restrict__ out) {
  int gid = blockIdx.x;
  int b  = gid / QTILES;
  int q0 = (gid % QTILES) * 16;
  int lane = threadIdx.x;
  int lp = lane & 15;
  int hi = (lane >= 16) ? 1 : 0;

  // Q as B-matrix (q^T): lane l holds q-row l%16, elements 0..7 = d 8*hi..+7; elements 8..15 = K pad (zero)
  v16bf qB = {};
  {
    const bf16_t* qr = qb + ((size_t)b*NQ_ + q0 + lp)*DK_ + hi*8;
    v8bf a = *(const v8bf*)qr;
#pragma unroll
    for (int i = 0; i < 8; i++) { qB[i] = a[i]; }
  }

  v8f O[8];
#pragma unroll
  for (int j = 0; j < 8; j++) O[j] = (v8f){};
  float m = -INFINITY, l = 0.0f;

#pragma unroll 1
  for (int blk = 0; blk < KB32; blk++) {
    int k0 = blk * 32;
    // K tiles as A-matrix: lane row = key, K = d (padded to 32)
    v16bf kA0 = {}, kA1 = {};
    {
      const bf16_t* r0 = kp + ((size_t)b*NK_ + k0 + lp)*DK_ + hi*8;
      const bf16_t* r1 = kp + ((size_t)b*NK_ + k0 + 16 + lp)*DK_ + hi*8;
      v8bf t0 = *(const v8bf*)r0;
      v8bf t1 = *(const v8bf*)r1;
#pragma unroll
      for (int i = 0; i < 8; i++) { kA0[i] = t0[i]; kA1[i] = t1[i]; }
    }
    // S^T tiles: lane = query column; VGPR r = key r (lo half) / r+8 (hi half)
    v8f s0 = wmma_bf16(kA0, qB, (v8f){});
    v8f s1 = wmma_bf16(kA1, qB, (v8f){});

    // online softmax (per query row = per lane; halves combined with shfl_xor 16)
    float lmax = -INFINITY;
#pragma unroll
    for (int r = 0; r < 8; r++) lmax = fmaxf(lmax, fmaxf(s0[r], s1[r]));
    float bmax = fmaxf(lmax, __shfl_xor(lmax, 16));
    float mnew = fmaxf(m, bmax);
    float scale = __expf(m - mnew);
    v8f p0, p1;
    float psum = 0.0f;
#pragma unroll
    for (int r = 0; r < 8; r++) {
      p0[r] = __expf(s0[r] - mnew);
      p1[r] = __expf(s1[r] - mnew);
      psum += p0[r] + p1[r];
    }
    l = l * scale + (psum + __shfl_xor(psum, 16));
    m = mnew;
#pragma unroll
    for (int j = 0; j < 8; j++)
#pragma unroll
      for (int r = 0; r < 8; r++) O[j][r] *= scale;

    // build P^T in B-layout: exchange halves so lane n holds keys 0..15, lane n+16 keys 16..31
    v16bf P;
#pragma unroll
    for (int r = 0; r < 8; r++) {
      float p0o = __shfl_xor(p0[r], 16);
      float p1o = __shfl_xor(p1[r], 16);
      P[r]     = (bf16_t)(hi ? p1o   : p0[r]);
      P[8 + r] = (bf16_t)(hi ? p1[r] : p0o);
    }

    // O^T += V^T x P^T  (8 channel chunks, K = 32 keys)
#pragma unroll
    for (int j = 0; j < 8; j++) {
      const bf16_t* vr = vt + ((size_t)b*C_ + j*16 + lp)*NK_ + k0 + hi*16;
      v8bf v0 = *(const v8bf*)vr;
      v8bf v1 = *(const v8bf*)(vr + 8);
      v16bf Va;
#pragma unroll
      for (int i = 0; i < 8; i++) { Va[i] = v0[i]; Va[8+i] = v1[i]; }
      O[j] = wmma_bf16(Va, P, O[j]);
    }
  }

  // epilogue: out = x + beta * O/l  (lane = query; VGPR r = channel j*16 + r (+8 hi half))
  float invl = 1.0f / l;
  float bet = beta[0];
  size_t rowbase = ((size_t)b*NQ_ + q0 + lp) * C_;
#pragma unroll
  for (int j = 0; j < 8; j++) {
    int c0 = j*16 + hi*8;
#pragma unroll
    for (int r = 0; r < 8; r++) {
      size_t idx = rowbase + c0 + r;
      out[idx] = query[idx] + bet * O[j][r] * invl;
    }
  }
}

// ============================== launcher ====================================
extern "C" void kernel_launch(void* const* d_in, const int* in_sizes, int n_in,
                              void* d_out, int out_size, void* d_ws, size_t ws_size,
                              hipStream_t stream) {
  const float* query = (const float*)d_in[0];
  const float* Wq    = (const float*)d_in[1];
  const float* bq    = (const float*)d_in[2];
  const float* Wk    = (const float*)d_in[3];
  const float* bk    = (const float*)d_in[4];
  const float* Wv    = (const float*)d_in[5];
  const float* bv    = (const float*)d_in[6];
  const float* beta  = (const float*)d_in[7];
  float* out = (float*)d_out;

  char* ws = (char*)d_ws;
  bf16_t* WqT = (bf16_t*)(ws + 0);                       //  4 KB
  bf16_t* WkT = (bf16_t*)(ws + 4096);                    //  4 KB
  bf16_t* WvT = (bf16_t*)(ws + 8192);                    // 32 KB
  bf16_t* qb  = (bf16_t*)(ws + 40960);                   // 8*9216*16*2  = 2359296 B
  bf16_t* kp  = (bf16_t*)(ws + 40960 + 2359296);         // 8*2304*16*2  =  589824 B
  bf16_t* vt  = (bf16_t*)(ws + 40960 + 2359296 + 589824);// 8*128*2304*2 = 4718592 B

  prep_weights_k<<<(C_*C_ + 255)/256, 256, 0, stream>>>(Wq, Wk, Wv, WqT, WkT, WvT);
  stageA_k<<<B_ * PH_ * (W_/16), 32, 0, stream>>>(query, bq, bk, bv, WqT, WkT, WvT, qb, kp, vt);
  stageB_k<<<B_ * QTILES, 32, 0, stream>>>(query, qb, kp, vt, beta, out);
}
